// BasicNCA2D_40484361732833
// MI455X (gfx1250) — matrix operand order repaired and checked by reading, not compiled, not run on hardware
//
#include <hip/hip_runtime.h>
#include <cstdint>

typedef __attribute__((ext_vector_type(16))) __bf16 v16bf;
typedef __attribute__((ext_vector_type(8)))  float  v8f;
typedef __attribute__((ext_vector_type(2)))  float  f32x2;

#define NB  8
#define NH  256
#define NW  256
#define NC  16
#define HID 128
#define TILE_H 8
#define TILE_W 16
#define HALO 3
#define XT_H (TILE_H + 2*HALO)   // 14
#define XT_W (TILE_W + 2*HALO)   // 22

union BVec   { uint4 u[2]; v16bf v; };      // 32B: one 16-bit WMMA fragment
union HChunk { uint4 u; __bf16 h[8]; };     // 16B: 8 bf16
union F8     { float4 f[2]; float s[8]; };  // 32B: 8 f32
union P2     { uint32_t u; __bf16 h[2]; };  // bf16 pair

__device__ __forceinline__ int reflect256(int i) {
    return i < 0 ? -i : (i > 255 ? 510 - i : i);
}

__device__ __forceinline__ uint32_t pcg_hash(uint32_t v) {
    v = v * 747796405u + 2891336453u;
    uint32_t w = ((v >> ((v >> 28u) + 4u)) ^ v) * 277803737u;
    return (w >> 22u) ^ w;
}

// low 32 bits of a flat shared pointer == LDS byte offset (aperture rule)
__device__ __forceinline__ uint32_t lds_off(const void* p) {
    return (uint32_t)(uintptr_t)p;
}

// ---------------------------------------------------------------------------
// Kernel 1: depthwise 7x7 conv (reflect pad, packed f32 pairs) + concat +
// FC0 (32->128) WMMA. Writes h (bf16, [cell][128]) via async LDS->global DMA,
// accumulates per-channel sum/sum^2 for BatchNorm batch statistics.
// Block: 256 threads = 8 waves; 8x16 spatial tile of one image.
// ---------------------------------------------------------------------------
__global__ __launch_bounds__(256) void nca_k1(
    const float* __restrict__ x, const float* __restrict__ pw,
    const float* __restrict__ pb, const float* __restrict__ w0,
    const float* __restrict__ b0, unsigned short* __restrict__ hout_raw,
    float* __restrict__ gSum, float* __restrict__ gSum2)
{
    __bf16* hout = (__bf16*)hout_raw;

    __shared__ alignas(16) union {
        float  xt[XT_H * XT_W * NC];          // 19712 B: x tile + halo
        __bf16 hl[TILE_H * TILE_W * HID];     // 32768 B: h tile (post-conv reuse)
    } u;
    __shared__ alignas(16) __bf16 yt[TILE_H * TILE_W * 32]; // perceive feats
    __shared__ alignas(16) __bf16 w0t[HID * 32];            // fc0 w, transposed [n][K]
    __shared__ alignas(8)  float  pwl[49 * NC];
    __shared__ alignas(8)  float  pbl[NC];
    __shared__ float  b0l[HID];
    __shared__ float  sS[HID], sS2[HID];

    const int t   = threadIdx.x;
    const int bid = blockIdx.x;
    const int tw  = bid & 15, th = (bid >> 4) & 31, b = bid >> 9;
    const int gr0 = th * TILE_H, gc0 = tw * TILE_W;

    // --- async DMA: stage x tile (reflect halo) straight into LDS ----------
    {
        const uint64_t xbase = (uint64_t)(uintptr_t)x;
        const uint32_t xtl   = lds_off(u.xt);
        for (int i = t; i < XT_H * XT_W * 4; i += 256) {  // 1232 x 16B chunks
            int pos = i >> 2, q = i & 3;
            int rr = pos / XT_W, cc = pos % XT_W;
            int gr = reflect256(gr0 + rr - HALO);
            int gc = reflect256(gc0 + cc - HALO);
            uint32_t voff = (uint32_t)((((((b << 8) + gr) << 8) + gc) * NC + q * 4) * 4);
            uint32_t dst  = xtl + (uint32_t)(pos * 64 + q * 16);
            asm volatile("global_load_async_to_lds_b128 %0, %1, %2"
                         :: "v"(dst), "v"(voff), "s"(xbase) : "memory");
        }
    }

    // --- stage weights (transposed fc0) while the DMA runs ----------------
    for (int i = t; i < 49 * NC; i += 256) pwl[i] = pw[i];
    for (int i = t; i < 32 * HID; i += 256) {
        int K = i >> 7, n = i & 127;                    // w0 is [32][128]
        w0t[n * 32 + K] = (__bf16)w0[i];
    }
    if (t < NC)  pbl[t] = pb[t];
    if (t < HID) { b0l[t] = b0[t]; sS[t] = 0.f; sS2[t] = 0.f; }

    asm volatile("s_wait_asynccnt 0x0" ::: "memory");
    __syncthreads();

    // --- depthwise 7x7 conv on channel pairs (v_pk_fma_f32 path) -----------
    {
        const f32x2* xt2 = (const f32x2*)u.xt;
        const f32x2* pw2 = (const f32x2*)pwl;
        const f32x2* pb2 = (const f32x2*)pbl;
        #pragma unroll
        for (int i = 0; i < 4; i++) {
            int e   = t + (i << 8);                      // 1024 f32x2 outputs
            int chp = e & 7, col = (e >> 3) & 15, row = e >> 7;
            f32x2 acc = pb2[chp];
            #pragma unroll
            for (int dy = 0; dy < 7; dy++)
                #pragma unroll
                for (int dx = 0; dx < 7; dx++)
                    acc = __builtin_elementwise_fma(
                              xt2[((row + dy) * XT_W + (col + dx)) * 8 + chp],
                              pw2[(dy * 7 + dx) * 8 + chp], acc);
            P2 pk; pk.h[0] = (__bf16)acc[0]; pk.h[1] = (__bf16)acc[1];
            *(uint32_t*)&yt[(row * 16 + col) * 32 + 16 + 2 * chp] = pk.u;
            f32x2 xv = xt2[((row + HALO) * XT_W + (col + HALO)) * 8 + chp];
            P2 px; px.h[0] = (__bf16)xv[0]; px.h[1] = (__bf16)xv[1];
            *(uint32_t*)&yt[(row * 16 + col) * 32 + 2 * chp] = px.u;
        }
    }
    __syncthreads();

    // --- FC0 via WMMA: A = 16 cells x 32 feats, 8 N-tiles of 16 ------------
    const int w = t >> 5, lane = t & 31;
    const int m = lane & 15, kh = lane >> 4;

    // A fragment: two contiguous 16B runs per lane (ISA 16-bit A layout)
    const uint4* ytv = (const uint4*)yt;
    BVec av;
    av.u[0] = ytv[(w * 16 + m) * 4 + kh];
    av.u[1] = ytv[(w * 16 + m) * 4 + 2 + kh];

    const uint4* w0v = (const uint4*)w0t;
    #pragma unroll
    for (int n = 0; n < 8; n++) {
        BVec bv;
        int bi = (n * 16 + m) * 4 + kh;                 // [col][K] transposed
        bv.u[0] = w0v[bi];
        bv.u[1] = w0v[bi + 2];
        // preload bias into the accumulator (lane owns one output channel)
        float bias = b0l[n * 16 + m];
        v8f Cc;
        #pragma unroll
        for (int r = 0; r < 8; r++) Cc[r] = bias;
        Cc = __builtin_amdgcn_wmma_f32_16x16x32_bf16(
                 false, av.v, false, bv.v, (short)0, Cc, false, false);
        float s = 0.f, s2 = 0.f;
        #pragma unroll
        for (int r = 0; r < 8; r++) {
            float d = Cc[r];
            int M = kh * 8 + r;                          // D layout M half
            u.hl[(w * 16 + M) * HID + n * 16 + m] = (__bf16)d;
            s += d; s2 += d * d;
        }
        atomicAdd(&sS[n * 16 + m], s);
        atomicAdd(&sS2[n * 16 + m], s2);
    }
    __syncthreads();

    // --- async DMA: h tile LDS -> global (each tile row 4KB contiguous) ----
    {
        const uint64_t hbase = (uint64_t)(uintptr_t)hout;
        const uint32_t hll   = lds_off(u.hl);
        #pragma unroll
        for (int idx = t; idx < 2048; idx += 256) {      // 2048 x 16B chunks
            int row = idx >> 8, c = idx & 255;
            uint32_t cellrow = (uint32_t)(((((b << 8) + (gr0 + row)) << 8) + gc0));
            uint32_t voff = cellrow * 256u + (uint32_t)c * 16u;  // bytes into hout
            uint32_t src  = hll + (uint32_t)idx * 16u;
            asm volatile("global_store_async_from_lds_b128 %0, %1, %2"
                         :: "v"(voff), "v"(src), "s"(hbase) : "memory");
        }
    }
    if (t < HID) {
        atomicAdd(&gSum[t],  sS[t]);
        atomicAdd(&gSum2[t], sS2[t]);
    }
    // implicit s_wait_idle at s_endpgm covers outstanding async stores
}

// ---------------------------------------------------------------------------
// Kernel 2: finalize BatchNorm batch statistics -> per-channel scale/shift
// ---------------------------------------------------------------------------
__global__ __launch_bounds__(128) void nca_k2(
    const float* __restrict__ gSum, const float* __restrict__ gSum2,
    const float* __restrict__ gamma, const float* __restrict__ beta,
    float* __restrict__ scale, float* __restrict__ shift)
{
    int t = threadIdx.x;
    const float invN = 1.0f / (float)((size_t)NB * NH * NW);
    float mean = gSum[t] * invN;
    float var  = gSum2[t] * invN - mean * mean;
    float inv  = rsqrtf(var + 1e-5f);
    float sc   = gamma[t] * inv;
    scale[t] = sc;
    shift[t] = beta[t] - mean * sc;
}

// ---------------------------------------------------------------------------
// Kernel 3: BN + ReLU + FC1 (128->16) chained WMMA + stochastic update.
// Block: 256 threads = 8 waves; 128 consecutive cells (32KB of h, contiguous,
// DMA'd into LDS via the async path).
// ---------------------------------------------------------------------------
__global__ __launch_bounds__(256) void nca_k3(
    const unsigned short* __restrict__ hin_raw, const float* __restrict__ xin,
    const float* __restrict__ w1, const float* __restrict__ scale,
    const float* __restrict__ shift, float* __restrict__ xout, int step)
{
    const __bf16* hin = (const __bf16*)hin_raw;

    __shared__ alignas(16) __bf16 hl[128 * HID];   // 32 KB
    __shared__ alignas(16) __bf16 w1t[NC * HID];   // 4 KB, transposed [col][K]
    __shared__ alignas(16) float  scl[HID], shf[HID];

    const int t  = threadIdx.x;
    const size_t cb = (size_t)blockIdx.x * 128;

    // --- async DMA: 32KB contiguous h tile -> LDS --------------------------
    {
        const uint64_t gbase = (uint64_t)(uintptr_t)(hin + cb * HID);
        const uint32_t hll   = lds_off(hl);
        #pragma unroll
        for (int i = 0; i < 8; i++) {
            uint32_t off = (uint32_t)(t * 16 + i * 4096);
            uint32_t dst = hll + off;
            asm volatile("global_load_async_to_lds_b128 %0, %1, %2"
                         :: "v"(dst), "v"(off), "s"(gbase) : "memory");
        }
    }
    for (int i = t; i < NC * HID; i += 256) {
        int K = i >> 4, col = i & 15;                   // w1 is [128][16]
        w1t[col * HID + K] = (__bf16)w1[i];
    }
    if (t < HID) { scl[t] = scale[t]; shf[t] = shift[t]; }

    asm volatile("s_wait_asynccnt 0x0" ::: "memory");
    __syncthreads();

    const int w = t >> 5, lane = t & 31;
    const int m = lane & 15, kh = lane >> 4;
    const int cellIdx = w * 16 + m;

    const uint4*   hlv  = (const uint4*)hl;
    const uint4*   w1v  = (const uint4*)w1t;
    const float4*  sclv = (const float4*)scl;
    const float4*  shfv = (const float4*)shf;

    v8f Cc = {};
    #pragma unroll
    for (int k = 0; k < 4; k++) {
        // two contiguous 16B h-chunks per lane (ch runs of 8)
        HChunk c0, c1;
        c0.u = hlv[cellIdx * 16 + k * 4 + kh];          // ch = k*32+kh*8 ..+7
        c1.u = hlv[cellIdx * 16 + k * 4 + kh + 2];      // ch = k*32+16+kh*8 ..+7
        F8 S0, S1, H0, H1;
        S0.f[0] = sclv[k * 8 + kh * 2];     S0.f[1] = sclv[k * 8 + kh * 2 + 1];
        H0.f[0] = shfv[k * 8 + kh * 2];     H0.f[1] = shfv[k * 8 + kh * 2 + 1];
        S1.f[0] = sclv[k * 8 + 4 + kh * 2]; S1.f[1] = sclv[k * 8 + 4 + kh * 2 + 1];
        H1.f[0] = shfv[k * 8 + 4 + kh * 2]; H1.f[1] = shfv[k * 8 + 4 + kh * 2 + 1];

        BVec av;
        #pragma unroll
        for (int ii = 0; ii < 8; ii++) {
            float v0 = (float)c0.h[ii] * S0.s[ii] + H0.s[ii];   // BatchNorm
            float v1 = (float)c1.h[ii] * S1.s[ii] + H1.s[ii];
            ((__bf16*)&av)[ii]     = (__bf16)fmaxf(v0, 0.0f);   // ReLU
            ((__bf16*)&av)[8 + ii] = (__bf16)fmaxf(v1, 0.0f);
        }
        BVec bv;
        int bi = m * 16 + k * 4 + kh;                    // [col][K] transposed
        bv.u[0] = w1v[bi];
        bv.u[1] = w1v[bi + 2];
        Cc = __builtin_amdgcn_wmma_f32_16x16x32_bf16(
                 false, av.v, false, bv.v, (short)0, Cc, false, false);
    }

    // --- stochastic residual update; channel 0 frozen ----------------------
    #pragma unroll
    for (int r = 0; r < 8; r++) {
        int M = kh * 8 + r;
        size_t cell = cb + (size_t)w * 16 + M;
        uint32_t h  = pcg_hash((uint32_t)cell ^ (0x9E3779B9u * (uint32_t)(step + 1)));
        float rnd   = (float)h * 2.3283064365386963e-10f;
        float mask  = rnd > 0.5f ? 1.0f : 0.0f;
        float xold  = xin[cell * NC + m];
        float outv  = (m == 0) ? xold : xold + Cc[r] * mask;
        xout[cell * NC + m] = outv;
    }
}

// ---------------------------------------------------------------------------
extern "C" void kernel_launch(void* const* d_in, const int* in_sizes, int n_in,
                              void* d_out, int out_size, void* d_ws, size_t ws_size,
                              hipStream_t stream) {
    (void)in_sizes; (void)n_in; (void)out_size; (void)ws_size;

    const float* x0   = (const float*)d_in[0];
    const float* p0w  = (const float*)d_in[1];
    const float* p0b  = (const float*)d_in[2];
    const float* fc0w = (const float*)d_in[3];
    const float* fc0b = (const float*)d_in[4];
    const float* gam  = (const float*)d_in[5];
    const float* bet  = (const float*)d_in[6];
    const float* fc1w = (const float*)d_in[7];
    // d_in[8] = steps: device scalar (=10); must be a capture-time constant
    // (no sync allowed during graph capture).
    const int STEPS = 10;

    const size_t H_BYTES = (size_t)NB * NH * NW * HID * 2;  // 128 MB bf16 h
    const size_t X_BYTES = (size_t)NB * NH * NW * NC * 4;   // 33.5 MB f32 x

    unsigned short* hbuf = (unsigned short*)d_ws;
    float* xA    = (float*)((char*)d_ws + H_BYTES);
    float* stats = (float*)((char*)d_ws + H_BYTES + X_BYTES);
    float* gSum  = stats;
    float* gSum2 = stats + 128;
    float* gScl  = stats + 256;
    float* gShf  = stats + 384;

    const int nBlocks = (NB * NH * NW) / (TILE_H * TILE_W);  // 4096

    const float* cur = x0;
    for (int s = 0; s < STEPS; s++) {
        float* nxt = (s & 1) ? (float*)d_out : xA;   // step 9 (odd) -> d_out
        hipMemsetAsync(gSum, 0, 256 * sizeof(float), stream);
        nca_k1<<<nBlocks, 256, 0, stream>>>(cur, p0w, p0b, fc0w, fc0b,
                                            hbuf, gSum, gSum2);
        nca_k2<<<1, 128, 0, stream>>>(gSum, gSum2, gam, bet, gScl, gShf);
        nca_k3<<<nBlocks, 256, 0, stream>>>(hbuf, cur, fc1w, gScl, gShf,
                                            nxt, s);
        cur = nxt;
    }
}
